// MultiHeadAttention_82016695484643
// MI455X (gfx1250) — matrix operand order repaired and checked by reading, not compiled, not run on hardware
//
#include <hip/hip_runtime.h>
#include <hip/hip_bf16.h>
#include <math.h>

typedef __attribute__((ext_vector_type(16))) __bf16 v16bf;
typedef __attribute__((ext_vector_type(8)))  float  v8f;

#define N_NODES 2048
#define IN_DIM  512
#define NHEAD   8
#define HDIM    64
#define LPATH   5
#define NEDGE   32768
#define D_EDGE  64
#define LH      (LPATH * NHEAD)   // 40

union frag16 {
  v16bf bf;
  unsigned int u[8];
};

// ---------------------------------------------------------------------------
// WMMA helpers (CDNA5 gfx1250, wave32). Layouts per CDNA5 ISA.
// ---------------------------------------------------------------------------
__device__ __forceinline__ v8f wmma_bf16(v16bf a, v16bf b, v8f c) {
  return __builtin_amdgcn_wmma_f32_16x16x32_bf16(
      /*neg_a=*/false, a, /*neg_b=*/false, b,
      /*c_mod=*/(short)0, c, /*reuse_a=*/false, /*reuse_b=*/false);
}

// A fragment 16x32 bf16 from row-major A (lda elems), rows row0..row0+15,
// k in [k0,k0+32). Lane (l&15)=M; l>>4 selects K interleave half.
// k-pairs are consecutive -> one dword load per pair.
__device__ __forceinline__ v16bf load_a_rowmajor(const __bf16* A, int lda,
                                                 int row0, int k0) {
  unsigned lane = threadIdx.x & 31u;
  unsigned M = lane & 15u, half = lane >> 4;
  const __bf16* base = A + (size_t)(row0 + M) * lda;
  frag16 f;
#pragma unroll
  for (int v = 0; v < 8; ++v) {
    int k = k0 + 2 * (v & 3) + 8 * (int)half + ((v >= 4) ? 16 : 0);
    f.u[v] = *(const unsigned int*)(base + k);
  }
  return f.bf;
}

// B fragment 32x16 bf16 where B[k][n] = S[(col0+n)*lds_ + k] (column-major /
// pre-transposed source). Lane (l&15)=N; l>>4 selects K interleave half
// (same k order as the A fragment). k-pairs consecutive -> dword loads.
__device__ __forceinline__ v16bf load_b_colmajor(const __bf16* S, int lds_,
                                                 int k0, int col0) {
  unsigned lane = threadIdx.x & 31u;
  unsigned n = lane & 15u;
  int kh = (int)(lane >> 4) * 8;
  const __bf16* base = S + (size_t)(col0 + n) * lds_ + k0 + kh;
  frag16 f;
#pragma unroll
  for (int v = 0; v < 8; ++v) f.u[v] = *(const unsigned int*)(base + 2 * (v & 3) + ((v >= 4) ? 16 : 0));
  return f.bf;
}

// ---------------------------------------------------------------------------
// f32 -> bf16 conversion (plain) and transpose variant (for weights)
// ---------------------------------------------------------------------------
__global__ void k_cvt_bf16(const float* __restrict__ src,
                           __bf16* __restrict__ dst, int n) {
  int i = blockIdx.x * blockDim.x + threadIdx.x;
  if (i < n) dst[i] = (__bf16)src[i];
}

// dst[c*R + r] = src[r*C + c]
__global__ void k_cvt_transpose_bf16(const float* __restrict__ src,
                                     __bf16* __restrict__ dst, int R, int C) {
  int i = blockIdx.x * blockDim.x + threadIdx.x;
  if (i >= R * C) return;
  int r = i / C, c = i % C;
  dst[(size_t)c * R + r] = (__bf16)src[i];
}

// ---------------------------------------------------------------------------
// Wave-tiled GEMM: out[M x N] = A_bf16[M x K] @ Bt_bf16[N x K]^T + bias
// (B passed pre-transposed: Bt[n][k]).  Each wave computes a 16x32 tile
// (one A frag -> two WMMAs).  block = 128 (4 waves), grid = (N/128, M/16).
// out_mode: 0 = f32 row-major, 1 = bf16 row-major, 2 = bf16 transposed
// ---------------------------------------------------------------------------
__device__ __forceinline__ void store_c_tile(v8f c, const float* bias,
                                             void* out, int out_mode, int M,
                                             int N, int row0, int col0) {
  unsigned lane = threadIdx.x & 31u;
  unsigned n = lane & 15u, half = lane >> 4;
#pragma unroll
  for (int i = 0; i < 8; ++i) {
    int Mr = row0 + i + 8 * (int)half;
    int Nc = col0 + (int)n;
    float v = c[i] + bias[Nc];
    if (out_mode == 0)
      ((float*)out)[(size_t)Mr * N + Nc] = v;
    else if (out_mode == 1)
      ((__bf16*)out)[(size_t)Mr * N + Nc] = (__bf16)v;
    else
      ((__bf16*)out)[(size_t)Nc * M + Mr] = (__bf16)v;
  }
}

__global__ void k_gemm_bf16(const __bf16* __restrict__ A,
                            const __bf16* __restrict__ Bt,
                            const float* __restrict__ bias,
                            void* __restrict__ out, int out_mode,
                            int M, int N, int Kd) {
  int wave = threadIdx.x >> 5;
  int row0 = blockIdx.y * 16;
  int col0 = blockIdx.x * 128 + wave * 32;
  if (row0 >= M || col0 >= N) return;
  v8f c0 = {}, c1 = {};
  for (int k0 = 0; k0 < Kd; k0 += 32) {
    v16bf a  = load_a_rowmajor(A, Kd, row0, k0);
    v16bf b0 = load_b_colmajor(Bt, Kd, k0, col0);
    v16bf b1 = load_b_colmajor(Bt, Kd, k0, col0 + 16);
    c0 = wmma_bf16(a, b0, c0);
    c1 = wmma_bf16(a, b1, c1);
  }
  store_c_tile(c0, bias, out, out_mode, M, N, row0, col0);
  store_c_tile(c1, bias, out, out_mode, M, N, row0, col0 + 16);
}

// ---------------------------------------------------------------------------
// T[e, l, h] = sum_d padded_edge_feat[e][d] * edge_weight[l*H+h][d]
// e in [0, E]; row E is the zero pad row -> T = 0.  T layout: e*40 + l*8 + h.
// ---------------------------------------------------------------------------
__global__ void k_edge_bias_table(const float* __restrict__ edge_feat,
                                  const float* __restrict__ edge_weight,
                                  float* __restrict__ T) {
  int idx = blockIdx.x * blockDim.x + threadIdx.x;
  const int total = (NEDGE + 1) * LH;
  if (idx >= total) return;
  int lh = idx % LH;
  int e  = idx / LH;
  float s = 0.f;
  if (e < NEDGE) {
    const float* ef = edge_feat + (size_t)e * D_EDGE;
    const float* w  = edge_weight + (size_t)lh * D_EDGE;
#pragma unroll 8
    for (int d = 0; d < D_EDGE; ++d) s += ef[d] * w[d];
  }
  T[idx] = s;
}

// ---------------------------------------------------------------------------
// Fused biased attention for one (16-row block, head).
// grid = (N/16, H), block = 256 (8 waves).
// Dynamic LDS: scores[16][2048] f32 (128KB) + partials 8*16*16 f32 (8KB).
// Vt is V pre-transposed: Vt[d_global][m], d_global = h*64 + d.
// ---------------------------------------------------------------------------
__global__ void k_attention(const __bf16* __restrict__ Qb,
                            const __bf16* __restrict__ Kb,
                            const __bf16* __restrict__ Vt,
                            const int* __restrict__ dist,
                            const int* __restrict__ paths,
                            const int* __restrict__ amask,
                            const float* __restrict__ sp_bias,
                            const float* __restrict__ T,
                            __bf16* __restrict__ AOut) {
  extern __shared__ float smem[];
  float* sc    = smem;                    // 16 * 2048
  float* parts = smem + 16 * N_NODES;     // 8 * 256

  const int nblk = blockIdx.x;
  const int h    = blockIdx.y;
  const int row0 = nblk * 16;
  const int wave = threadIdx.x >> 5;
  const unsigned lane = threadIdx.x & 31u;
  const unsigned nn = lane & 15u, half = lane >> 4;

  // ---- Phase 1: scores = (Q K^T)/8 + spatial + path bias, mask ----
  const v16bf aq0 = load_a_rowmajor(Qb + h * HDIM, IN_DIM, row0, 0);
  const v16bf aq1 = load_a_rowmajor(Qb + h * HDIM, IN_DIM, row0, 32);

  for (int t = wave; t < N_NODES / 16; t += 8) {
    int m0 = t * 16;
    // Prefetch the next tile's shortest-path rows (dominant HBM stream).
    if (t + 8 < N_NODES / 16) {
      int mp = (t + 8) * 16 + (int)nn;
#pragma unroll
      for (int i = 0; i < 8; ++i) {
        int gn = row0 + i + 8 * (int)half;
        __builtin_prefetch(paths + ((size_t)gn * N_NODES + mp) * LPATH, 0, 1);
      }
    }
    v16bf bk0 = load_b_colmajor(Kb + h * HDIM, IN_DIM, 0, m0);
    v16bf bk1 = load_b_colmajor(Kb + h * HDIM, IN_DIM, 32, m0);
    v8f c = {};
    c = wmma_bf16(aq0, bk0, c);
    c = wmma_bf16(aq1, bk1, c);
#pragma unroll
    for (int i = 0; i < 8; ++i) {
      int Mr = i + 8 * (int)half;
      int gn = row0 + Mr;
      int gm = m0 + (int)nn;
      float s = c[i] * 0.125f;  // 1/sqrt(64)
      int d = dist[(size_t)gn * N_NODES + gm];
      d = d < 0 ? 0 : (d > LPATH ? LPATH : d);
      s += sp_bias[d * NHEAD + h];
      const int* p = paths + ((size_t)gn * N_NODES + gm) * LPATH;
#pragma unroll
      for (int l = 0; l < LPATH; ++l) {
        int e = p[l];  // 0..E; E -> zero pad row of T
        s += T[(size_t)e * LH + l * NHEAD + h];
      }
      if (amask[(size_t)gn * N_NODES + gm] != 0) s = -INFINITY;
      sc[Mr * N_NODES + gm] = s;
    }
  }
  __syncthreads();

  // ---- Phase 2: softmax over the 2048 columns, 2 rows per wave ----
  for (int r = wave * 2; r < wave * 2 + 2; ++r) {
    float* row = sc + (size_t)r * N_NODES;
    float mx = -INFINITY;
    for (int j = (int)lane; j < N_NODES; j += 32) mx = fmaxf(mx, row[j]);
#pragma unroll
    for (int o = 16; o >= 1; o >>= 1) mx = fmaxf(mx, __shfl_xor(mx, o, 32));
    float sum = 0.f;
    for (int j = (int)lane; j < N_NODES; j += 32) {
      float e = __expf(row[j] - mx);
      row[j] = e;
      sum += e;
    }
#pragma unroll
    for (int o = 16; o >= 1; o >>= 1) sum += __shfl_xor(sum, o, 32);
    float inv = 1.0f / sum;
    for (int j = (int)lane; j < N_NODES; j += 32) row[j] *= inv;
  }
  __syncthreads();

  // ---- Phase 3: out(16x64) = P(16x2048) @ V(2048x64) ----
  // wave -> (ntile = wave&3 of 4 col-tiles, khalf = wave>>2 of 2 k-halves)
  const int ntile = wave & 3;
  const int khalf = wave >> 2;
  const int col0  = ntile * 16;
  v8f acc = {};
  const float2* pbase =
      (const float2*)(sc + (size_t)(lane & 15u) * N_NODES);
  for (int kk = 0; kk < 32; ++kk) {
    int k0 = khalf * 1024 + kk * 32;
    v16bf a;
#pragma unroll
    for (int v = 0; v < 8; ++v) {
      int k = k0 + 2 * (v & 3) + 8 * (int)half + ((v >= 4) ? 16 : 0);
      float2 pv = pbase[k >> 1];          // ds_load_b64
      a[2 * v]     = (__bf16)pv.x;
      a[2 * v + 1] = (__bf16)pv.y;
    }
    v16bf b =
        load_b_colmajor(Vt + (size_t)h * HDIM * N_NODES, N_NODES, k0, col0);
    acc = wmma_bf16(a, b, acc);
  }
  float* myp = parts + wave * 256;
#pragma unroll
  for (int i = 0; i < 8; ++i) myp[(i + 8 * (int)half) * 16 + nn] = acc[i];
  __syncthreads();
  if (wave < 4) {
#pragma unroll
    for (int i = 0; i < 8; ++i) {
      int Mr = i + 8 * (int)half;
      float v = parts[wave * 256 + Mr * 16 + nn] +
                parts[(wave + 4) * 256 + Mr * 16 + nn];
      AOut[(size_t)(row0 + Mr) * IN_DIM + h * HDIM + wave * 16 + nn] =
          (__bf16)v;
    }
  }
}

// ---------------------------------------------------------------------------
// Host launcher
// ---------------------------------------------------------------------------
extern "C" void kernel_launch(void* const* d_in, const int* in_sizes, int n_in,
                              void* d_out, int out_size, void* d_ws,
                              size_t ws_size, hipStream_t stream) {
  const float* node_feat  = (const float*)d_in[0];
  const int*   dist       = (const int*)d_in[3];
  const int*   paths      = (const int*)d_in[4];
  const float* edge_feat  = (const float*)d_in[5];
  const int*   amask      = (const int*)d_in[6];
  const float* WQ = (const float*)d_in[7];
  const float* bQ = (const float*)d_in[8];
  const float* WK = (const float*)d_in[9];
  const float* bK = (const float*)d_in[10];
  const float* WV = (const float*)d_in[11];
  const float* bV = (const float*)d_in[12];
  const float* WO = (const float*)d_in[13];
  const float* bO = (const float*)d_in[14];
  const float* sp_bias     = (const float*)d_in[15];
  const float* edge_weight = (const float*)d_in[16];

  char* ws = (char*)d_ws;
  size_t off = 0;
  auto alloc = [&](size_t bytes) {
    void* p = ws + off;
    off = (off + bytes + 255) & ~(size_t)255;
    return p;
  };
  __bf16* nf_b  = (__bf16*)alloc((size_t)N_NODES * IN_DIM * 2);
  __bf16* wqt_b = (__bf16*)alloc((size_t)IN_DIM * IN_DIM * 2);  // W^T bf16
  __bf16* wkt_b = (__bf16*)alloc((size_t)IN_DIM * IN_DIM * 2);
  __bf16* wvt_b = (__bf16*)alloc((size_t)IN_DIM * IN_DIM * 2);
  __bf16* wot_b = (__bf16*)alloc((size_t)IN_DIM * IN_DIM * 2);
  __bf16* Qb    = (__bf16*)alloc((size_t)N_NODES * IN_DIM * 2);
  __bf16* Kb    = (__bf16*)alloc((size_t)N_NODES * IN_DIM * 2);
  __bf16* Vt    = (__bf16*)alloc((size_t)N_NODES * IN_DIM * 2);  // transposed
  float*  T     = (float*)alloc((size_t)(NEDGE + 1) * LH * 4);
  __bf16* AOb   = (__bf16*)alloc((size_t)N_NODES * IN_DIM * 2);
  (void)ws_size; (void)in_sizes; (void)n_in; (void)out_size;

  // 1) conversions: node features plain, weights transposed
  {
    int n = N_NODES * IN_DIM;
    k_cvt_bf16<<<(n + 255) / 256, 256, 0, stream>>>(node_feat, nf_b, n);
    n = IN_DIM * IN_DIM;
    int g = (n + 255) / 256;
    k_cvt_transpose_bf16<<<g, 256, 0, stream>>>(WQ, wqt_b, IN_DIM, IN_DIM);
    k_cvt_transpose_bf16<<<g, 256, 0, stream>>>(WK, wkt_b, IN_DIM, IN_DIM);
    k_cvt_transpose_bf16<<<g, 256, 0, stream>>>(WV, wvt_b, IN_DIM, IN_DIM);
    k_cvt_transpose_bf16<<<g, 256, 0, stream>>>(WO, wot_b, IN_DIM, IN_DIM);
  }

  // 2) Q/K/V projections (Q,K bf16 row-major; V bf16 transposed)
  {
    dim3 grid(IN_DIM / 128, N_NODES / 16);
    k_gemm_bf16<<<grid, 128, 0, stream>>>(nf_b, wqt_b, bQ, Qb, 1, N_NODES,
                                          IN_DIM, IN_DIM);
    k_gemm_bf16<<<grid, 128, 0, stream>>>(nf_b, wkt_b, bK, Kb, 1, N_NODES,
                                          IN_DIM, IN_DIM);
    k_gemm_bf16<<<grid, 128, 0, stream>>>(nf_b, wvt_b, bV, Vt, 2, N_NODES,
                                          IN_DIM, IN_DIM);
  }

  // 3) path-edge bias table T (5.2 MB, L2-resident)
  {
    int total = (NEDGE + 1) * LH;
    k_edge_bias_table<<<(total + 255) / 256, 256, 0, stream>>>(
        edge_feat, edge_weight, T);
  }

  // 4) fused biased attention (scores staged in 136 KB dynamic LDS)
  {
    dim3 grid(N_NODES / 16, NHEAD);
    size_t shmem = (size_t)(16 * N_NODES + 8 * 256) * sizeof(float);
    k_attention<<<grid, 256, shmem, stream>>>(Qb, Kb, Vt, dist, paths, amask,
                                              sp_bias, T, AOb);
  }

  // 5) final output projection (f32 to d_out)
  {
    dim3 grid(IN_DIM / 128, N_NODES / 16);
    k_gemm_bf16<<<grid, 128, 0, stream>>>(AOb, wot_b, bO, d_out, 0, N_NODES,
                                          IN_DIM, IN_DIM);
  }
}
